// GCNClassifier_78288663871794
// MI455X (gfx1250) — matrix operand order repaired and checked by reading, not compiled, not run on hardware
//
#include <hip/hip_runtime.h>
#include <hip/hip_bf16.h>

#define GN 16384
#define DH 64

typedef __attribute__((ext_vector_type(16))) __bf16 v16bf;
typedef __attribute__((ext_vector_type(8)))  float  v8f;

union BF16Tile { v16bf v; unsigned int u[8]; };

__device__ __forceinline__ unsigned int pkbf(float a, float b) {
    // pack two f32 -> two bf16 (truncate); low half = a, high half = b
    return (__float_as_uint(a) >> 16) | (__float_as_uint(b) & 0xffff0000u);
}

// ---------------------------------------------------------------------------
// Big propagation GEMM: H[m][c] = relu( sum_k adj[m][k] * X[k][c] + bias[c] )
// adj: f32 [GN][GN] streamed from HBM, converted to bf16 in registers.
// XT : bf16 [DH][GN]  (X transposed -> contiguous K for the WMMA B operand)
// One wave owns 32 rows: 2 M-tiles x 4 N-tiles accumulators (all 64 cols).
// ---------------------------------------------------------------------------
__global__ __launch_bounds__(128) void gcn_prop_kernel(
    const float* __restrict__ adj,
    const unsigned short* __restrict__ XT,   // bf16 bits
    const float* __restrict__ bias,
    float* __restrict__ H)
{
    const int lane  = threadIdx.x & 31;
    const int wave  = threadIdx.x >> 5;
    const int mrow  = lane & 15;
    const int khalf = lane >> 4;                 // 0 or 1
    const int row0  = blockIdx.x * 128 + wave * 32;

    v8f acc[2][4];
#pragma unroll
    for (int mi = 0; mi < 2; ++mi)
#pragma unroll
        for (int ni = 0; ni < 4; ++ni)
            acc[mi][ni] = (v8f){0.f,0.f,0.f,0.f,0.f,0.f,0.f,0.f};

    // A-operand per-lane base: row (row0 + mi*16 + mrow), K offset khalf*8
    const float* aBase0 = adj + (size_t)(row0 +  0 + mrow) * GN + khalf * 8;
    const float* aBase1 = adj + (size_t)(row0 + 16 + mrow) * GN + khalf * 8;
    // B-operand per-lane base: column (ni*16 + mrow), K offset khalf*8
    const unsigned short* bBase[4];
#pragma unroll
    for (int ni = 0; ni < 4; ++ni)
        bBase[ni] = XT + (size_t)(ni * 16 + mrow) * GN + khalf * 8;

    for (int k = 0; k < GN; k += 32) {
        // ---- load + convert A tiles (adj f32 -> bf16, documented A layout) ----
        BF16Tile A[2];
#pragma unroll
        for (int mi = 0; mi < 2; ++mi) {
            const float4* p = (const float4*)((mi == 0 ? aBase0 : aBase1) + k);
            float4 c0 = p[0];   // K+0..3   (within this lane's K window)
            float4 c1 = p[1];   // K+4..7
            float4 c2 = p[4];   // K+16..19
            float4 c3 = p[5];   // K+20..23
            A[mi].u[0] = pkbf(c0.x, c0.y); A[mi].u[1] = pkbf(c0.z, c0.w);
            A[mi].u[2] = pkbf(c1.x, c1.y); A[mi].u[3] = pkbf(c1.z, c1.w);
            A[mi].u[4] = pkbf(c2.x, c2.y); A[mi].u[5] = pkbf(c2.z, c2.w);
            A[mi].u[6] = pkbf(c3.x, c3.y); A[mi].u[7] = pkbf(c3.z, c3.w);
        }
        // ---- load B tiles (already bf16, two 16B chunks per lane: K+0..7, K+16..23) ----
        BF16Tile B[4];
#pragma unroll
        for (int ni = 0; ni < 4; ++ni) {
            const uint4* q = (const uint4*)(bBase[ni] + k);
            uint4 q0 = q[0], q1 = q[2];
            B[ni].u[0] = q0.x; B[ni].u[1] = q0.y; B[ni].u[2] = q0.z; B[ni].u[3] = q0.w;
            B[ni].u[4] = q1.x; B[ni].u[5] = q1.y; B[ni].u[6] = q1.z; B[ni].u[7] = q1.w;
        }
        // prefetch next adj chunk into cache hierarchy (global_prefetch_b8)
        if (k + 32 < GN) {
            __builtin_prefetch(aBase0 + k + 32, 0, 1);
            __builtin_prefetch(aBase1 + k + 32, 0, 1);
        }
        // ---- 8x v_wmma_f32_16x16x32_bf16 ----
#pragma unroll
        for (int mi = 0; mi < 2; ++mi)
#pragma unroll
            for (int ni = 0; ni < 4; ++ni)
                acc[mi][ni] = __builtin_amdgcn_wmma_f32_16x16x32_bf16(
                    false, A[mi].v, false, B[ni].v,
                    (short)0, acc[mi][ni], false, false);
    }

    // ---- epilogue: bias + relu, store f32 (C/D layout: VGPR r -> M=r+khalf*8) ----
#pragma unroll
    for (int mi = 0; mi < 2; ++mi) {
#pragma unroll
        for (int ni = 0; ni < 4; ++ni) {
            const int col = ni * 16 + mrow;
            const float bv = bias[col];
#pragma unroll
            for (int r = 0; r < 8; ++r) {
                const int row = row0 + mi * 16 + khalf * 8 + r;
                float v = acc[mi][ni][r] + bv;
                H[(size_t)row * DH + col] = v > 0.f ? v : 0.f;
            }
        }
    }
}

// ---------------------------------------------------------------------------
// Small GEMM producing a transposed bf16 result: XT[j][n] = sum_k X[n][k]*W[k][j]
// ---------------------------------------------------------------------------
__global__ void trans_gemm_kernel(const float* __restrict__ X,
                                  const float* __restrict__ W,
                                  unsigned short* __restrict__ XT)
{
    int idx = blockIdx.x * blockDim.x + threadIdx.x;
    int n = idx >> 6, j = idx & 63;
    float acc = 0.f;
#pragma unroll
    for (int kk = 0; kk < DH; ++kk)
        acc += X[(size_t)n * DH + kk] * W[kk * DH + j];
    XT[(size_t)j * GN + n] = (unsigned short)(__float_as_uint(acc) >> 16);
}

// MLP head layer 1: H3 = relu(H @ Wm1 + bm1)
__global__ void head1_kernel(const float* __restrict__ H,
                             const float* __restrict__ Wm1,
                             const float* __restrict__ bm1,
                             float* __restrict__ H3)
{
    int idx = blockIdx.x * blockDim.x + threadIdx.x;
    int n = idx >> 6, j = idx & 63;
    float acc = bm1[j];
#pragma unroll
    for (int kk = 0; kk < DH; ++kk)
        acc += H[(size_t)n * DH + kk] * Wm1[kk * DH + j];
    H3[idx] = acc > 0.f ? acc : 0.f;
}

// MLP head layer 2: out = H3 @ Wm2 + bm2   ([N,1])
__global__ void head2_kernel(const float* __restrict__ H3,
                             const float* __restrict__ Wm2,
                             const float* __restrict__ bm2,
                             float* __restrict__ out)
{
    int n = blockIdx.x * blockDim.x + threadIdx.x;
    float acc = bm2[0];
#pragma unroll
    for (int kk = 0; kk < DH; ++kk)
        acc += H3[(size_t)n * DH + kk] * Wm2[kk];
    out[n] = acc;
}

extern "C" void kernel_launch(void* const* d_in, const int* in_sizes, int n_in,
                              void* d_out, int out_size, void* d_ws, size_t ws_size,
                              hipStream_t stream) {
    (void)in_sizes; (void)n_in; (void)out_size; (void)ws_size;
    const float* adj  = (const float*)d_in[0];
    const float* feat = (const float*)d_in[1];
    const float* W1   = (const float*)d_in[2];
    const float* b1   = (const float*)d_in[3];
    const float* W2   = (const float*)d_in[4];
    const float* b2   = (const float*)d_in[5];
    const float* Wm1  = (const float*)d_in[6];
    const float* bm1  = (const float*)d_in[7];
    const float* Wm2  = (const float*)d_in[8];
    const float* bm2  = (const float*)d_in[9];
    float* out = (float*)d_out;

    // workspace: XT (bf16, 2 MiB) | H (f32, 4 MiB) | H3 (f32, 4 MiB)
    unsigned short* XT = (unsigned short*)d_ws;
    float* H  = (float*)((char*)d_ws + (2u << 20));
    float* H3 = (float*)((char*)d_ws + (6u << 20));

    const int smallBlocks = (GN * DH) / 256;

    // Layer 1: XT1 = (features @ W1)^T ; H = relu(adj @ X1 + b1)
    trans_gemm_kernel<<<smallBlocks, 256, 0, stream>>>(feat, W1, XT);
    gcn_prop_kernel<<<GN / 128, 128, 0, stream>>>(adj, XT, b1, H);

    // Layer 2: XT2 = (H @ W2)^T ; H = relu(adj @ X2 + b2)
    trans_gemm_kernel<<<smallBlocks, 256, 0, stream>>>(H, W2, XT);
    gcn_prop_kernel<<<GN / 128, 128, 0, stream>>>(adj, XT, b2, H);

    // MLP head
    head1_kernel<<<smallBlocks, 256, 0, stream>>>(H, Wm1, bm1, H3);
    head2_kernel<<<GN / 256, 256, 0, stream>>>(H3, Wm2, bm2, out);
}
